// Block_29721173688983
// MI455X (gfx1250) — compile-verified, matrix operand order checked by reading, not run on hardware
//
#include <hip/hip_runtime.h>
#include <hip/hip_bf16.h>
#include <math.h>

// ---------------- problem constants (b=1) ----------------
#define LSEQ 2048   // sequence length
#define DMOD 1024   // d
#define DIN  2048   // 2*d inner channels
#define DTRK 64     // dt_rank
#define NST  16     // d_state
#define KCONV 4
#define SEGS 8
#define SEGLEN (LSEQ / SEGS)   // 256
#define CN16 (DIN * NST)       // 32768 (c,n) lanes

typedef __bf16 bf16x8  __attribute__((ext_vector_type(8)));
typedef __bf16 bf16x16 __attribute__((ext_vector_type(16)));
typedef float  f32x8   __attribute__((ext_vector_type(8)));

// ---------------- helpers ----------------
__device__ __forceinline__ float silu_f(float v) {
    return v * (1.0f / (1.0f + __expf(-v)));
}
__device__ __forceinline__ float softplus_f(float v) {
    return (v > 20.0f) ? v : log1pf(__expf(v));
}
// flat address of an LDS object: low 32 bits == LDS byte offset (LDS aperture)
__device__ __forceinline__ unsigned lds_off(const void* p) {
    return (unsigned)(unsigned long long)p;
}
// async global->LDS copy of 16 bytes per lane, tracked by ASYNCcnt
__device__ __forceinline__ void async_copy_b128(unsigned lds_byte_off, const void* gsrc) {
    unsigned long long src = (unsigned long long)gsrc;
    asm volatile("global_load_async_to_lds_b128 %0, %1, off"
                 :: "v"(lds_byte_off), "v"(src) : "memory");
}
__device__ __forceinline__ void async_wait0() {
    asm volatile("s_wait_asynccnt 0x0" ::: "memory");
}

// Load one 16x32 bf16 A/B fragment half-row: lane holds K = {kh*8+0..7, 16+kh*8+0..7}
// base already points at (row * K + k0 + khalf*8)
__device__ __forceinline__ bf16x16 load_frag(const __bf16* base) {
    bf16x8 lo = *(const bf16x8*)(base);
    bf16x8 hi = *(const bf16x8*)(base + 16);
    bf16x16 r;
#pragma unroll
    for (int i = 0; i < 8; ++i) { r[i] = lo[i]; r[i + 8] = hi[i]; }
    return r;
}

// ---------------- weight prepack: transpose f32(RxC) -> bf16(CxR) ----------------
__global__ void transpose_to_bf16(const float* __restrict__ in, __bf16* __restrict__ out,
                                  int R, int C) {
    int idx = blockIdx.x * blockDim.x + threadIdx.x;
    if (idx >= R * C) return;
    int r = idx / C, c = idx - r * C;
    out[(size_t)c * R + r] = (__bf16)in[idx];
}

// ---------------- RMSNorm -> bf16 ----------------
__global__ void rmsnorm_kernel(const float* __restrict__ x, const float* __restrict__ scale,
                               __bf16* __restrict__ h) {
    __shared__ float red[8];
    int row = blockIdx.x;
    const float* xr = x + (size_t)row * DMOD;
    float s = 0.f;
    for (int i = threadIdx.x; i < DMOD; i += blockDim.x) { float v = xr[i]; s += v * v; }
#pragma unroll
    for (int off = 16; off > 0; off >>= 1) s += __shfl_xor(s, off, 32);
    if ((threadIdx.x & 31) == 0) red[threadIdx.x >> 5] = s;
    __syncthreads();
    float tot = 0.f;
#pragma unroll
    for (int w = 0; w < 8; ++w) tot += red[w];
    float rstd = rsqrtf(tot / (float)DMOD + 1e-5f);
    for (int i = threadIdx.x; i < DMOD; i += blockDim.x)
        h[(size_t)row * DMOD + i] = (__bf16)(xr[i] * rstd * scale[i]);
}

// ---------------- epilogue (shared by both GEMM kernels) ----------------
// EPI 0: plain f32 store
// EPI 1: split halves: col<N/2 -> out0 raw ; col>=N/2 -> out1 silu
// EPI 2: out0 = softplus(acc + aux[col])
// EPI 3: out0 = aux[row*N+col] + acc  (residual)
template <int EPI>
__device__ __forceinline__ void epi_store(float v, int row, int col, int N,
                                          float* out0, float* out1, const float* aux) {
    if (EPI == 0) {
        out0[(size_t)row * N + col] = v;
    } else if (EPI == 1) {
        int h = N >> 1;
        if (col < h) out0[(size_t)row * h + col] = v;
        else         out1[(size_t)row * h + (col - h)] = silu_f(v);
    } else if (EPI == 2) {
        out0[(size_t)row * N + col] = softplus_f(v + aux[col]);
    } else {
        out0[(size_t)row * N + col] = aux[(size_t)row * N + col] + v;
    }
}

// ---------------- big-tile WMMA GEMM: 32x64 per wave, 64x128 per block -------------
// C(MxN) = A(MxK bf16 rowmajor) x Bt(NxK bf16 rowmajor); M%64==0, N%128==0, K%32==0
template <int EPI>
__global__ void __launch_bounds__(128)
gemm_big_wmma(const __bf16* __restrict__ A, const __bf16* __restrict__ Bt,
              int M, int N, int Kd,
              float* __restrict__ out0, float* __restrict__ out1,
              const float* __restrict__ aux) {
    const int lane  = threadIdx.x & 31;
    const int wave  = threadIdx.x >> 5;
    const int wm    = wave >> 1, wn = wave & 1;
    const int m0    = blockIdx.y * 64 + wm * 32;
    const int n0    = blockIdx.x * 128 + wn * 64;
    const int r     = lane & 15;
    const int khalf = lane >> 4;

    const __bf16* arow0 = A + (size_t)(m0 + r) * Kd + khalf * 8;
    const __bf16* arow1 = arow0 + (size_t)16 * Kd;
    const __bf16* bc[4];
#pragma unroll
    for (int j = 0; j < 4; ++j) bc[j] = Bt + (size_t)(n0 + j * 16 + r) * Kd + khalf * 8;

    f32x8 acc[2][4] = {};
    for (int k = 0; k < Kd; k += 32) {
        __builtin_prefetch((const void*)(arow0 + k + 256), 0, 3);
        __builtin_prefetch((const void*)(arow1 + k + 256), 0, 3);
#pragma unroll
        for (int j = 0; j < 4; ++j) __builtin_prefetch((const void*)(bc[j] + k + 256), 0, 3);
        bf16x16 a0 = load_frag(arow0 + k);
        bf16x16 a1 = load_frag(arow1 + k);
#pragma unroll
        for (int j = 0; j < 4; ++j) {
            bf16x16 b = load_frag(bc[j] + k);
            acc[0][j] = __builtin_amdgcn_wmma_f32_16x16x32_bf16(false, a0, false, b, (short)0, acc[0][j], false, false);
            acc[1][j] = __builtin_amdgcn_wmma_f32_16x16x32_bf16(false, a1, false, b, (short)0, acc[1][j], false, false);
        }
    }

    // D layout: vgpr rr -> M = tile_m + rr + khalf*8 ; N = tile_n + (lane&15)
#pragma unroll
    for (int i = 0; i < 2; ++i) {
        int rowbase = m0 + i * 16 + khalf * 8;
#pragma unroll
        for (int j = 0; j < 4; ++j) {
            int col = n0 + j * 16 + r;
#pragma unroll
            for (int rr = 0; rr < 8; ++rr)
                epi_store<EPI>(acc[i][j][rr], rowbase + rr, col, N, out0, out1, aux);
        }
    }
}

// ---------------- small-tile WMMA GEMM (16x32 per wave) for N=96 ----------------
template <int EPI>
__global__ void gemm_bf16_wmma(const __bf16* __restrict__ A, const __bf16* __restrict__ Bt,
                               int M, int N, int Kd,
                               float* __restrict__ out0, float* __restrict__ out1,
                               const float* __restrict__ aux) {
    const int lane  = threadIdx.x & 31;
    const int wave  = threadIdx.x >> 5;            // 0..3, stacked in M
    const int m0    = (blockIdx.y * 4 + wave) * 16;
    const int n0    = blockIdx.x * 32;
    const int r     = lane & 15;
    const int khalf = lane >> 4;

    const __bf16* arow  = A  + (size_t)(m0 + r)      * Kd + khalf * 8;
    const __bf16* bcol0 = Bt + (size_t)(n0 + r)      * Kd + khalf * 8;
    const __bf16* bcol1 = Bt + (size_t)(n0 + 16 + r) * Kd + khalf * 8;

    f32x8 acc0 = {}; f32x8 acc1 = {};
    for (int k = 0; k < Kd; k += 32) {
        __builtin_prefetch((const void*)(arow  + k + 128), 0, 3);
        __builtin_prefetch((const void*)(bcol0 + k + 128), 0, 3);
        __builtin_prefetch((const void*)(bcol1 + k + 128), 0, 3);
        bf16x16 a  = load_frag(arow  + k);
        bf16x16 b0 = load_frag(bcol0 + k);
        bf16x16 b1 = load_frag(bcol1 + k);
        acc0 = __builtin_amdgcn_wmma_f32_16x16x32_bf16(false, a, false, b0, (short)0, acc0, false, false);
        acc1 = __builtin_amdgcn_wmma_f32_16x16x32_bf16(false, a, false, b1, (short)0, acc1, false, false);
    }
    const int rowbase = m0 + khalf * 8;
    const int c0 = n0 + r;
#pragma unroll
    for (int rr = 0; rr < 8; ++rr) {
        epi_store<EPI>(acc0[rr], rowbase + rr, c0,      N, out0, out1, aux);
        epi_store<EPI>(acc1[rr], rowbase + rr, c0 + 16, N, out0, out1, aux);
    }
}

// ---------------- causal depthwise conv (K=4) + SiLU ----------------
__global__ void conv_silu_kernel(const float* __restrict__ xs_pre, const float* __restrict__ cw,
                                 const float* __restrict__ cb, float* __restrict__ xs_conv,
                                 __bf16* __restrict__ xs_bf) {
    int idx = blockIdx.x * blockDim.x + threadIdx.x;
    if (idx >= LSEQ * DIN) return;
    int l = idx / DIN, c = idx - l * DIN;
    float acc = cb[c];
#pragma unroll
    for (int k = 0; k < KCONV; ++k) {
        int ll = l - (KCONV - 1) + k;
        if (ll >= 0) acc += xs_pre[(size_t)ll * DIN + c] * cw[k * DIN + c];
    }
    float v = silu_f(acc);
    xs_conv[idx] = v;
    xs_bf[idx]   = (__bf16)v;
}

// ---------------- split proj -> deltaLow(bf16), B, C ----------------
__global__ void split_proj_kernel(const float* __restrict__ proj, __bf16* __restrict__ dlow,
                                  float* __restrict__ Bm, float* __restrict__ Cm) {
    int idx = blockIdx.x * blockDim.x + threadIdx.x;
    if (idx >= LSEQ * (DTRK + 2 * NST)) return;
    int l = idx / (DTRK + 2 * NST), j = idx - l * (DTRK + 2 * NST);
    float v = proj[idx];
    if (j < DTRK)            dlow[(size_t)l * DTRK + j] = (__bf16)v;
    else if (j < DTRK + NST) Bm[(size_t)l * NST + (j - DTRK)] = v;
    else                     Cm[(size_t)l * NST + (j - DTRK - NST)] = v;
}

// ---------------- selective scan, phase 1: per-segment carries ----------------
// block = 256 threads = 16 channels x 16 states; one segment of 256 steps per block.
// Stage delta/xs (16 channels) and B rows for the segment into LDS with async copies.
__global__ void __launch_bounds__(256)
scan_part1(const float* __restrict__ delta, const float* __restrict__ xsc,
           const float* __restrict__ Alog, const float* __restrict__ Bm,
           float* __restrict__ aCar, float* __restrict__ bCar) {
    __shared__ float shB[SEGLEN * NST];   // 16 KB
    __shared__ float shD[SEGLEN * 16];    // 16 KB
    __shared__ float shX[SEGLEN * 16];    // 16 KB
    const int t    = threadIdx.x;
    const int blk  = blockIdx.x;
    const int seg  = blk >> 7;            // 128 blocks per segment
    const int cblk = blk & 127;
    const int c0   = cblk * 16;
    const int l0   = seg * SEGLEN;

    const unsigned dB = lds_off(shB), dD = lds_off(shD), dX = lds_off(shX);
    const float* gB = Bm + (size_t)l0 * NST;
#pragma unroll
    for (int j = 0; j < 4; ++j) {
        int e = j * 256 + t;              // b128 chunk index
        async_copy_b128(dB + e * 16, gB + e * 4);
        int l = e >> 2, q = e & 3;
        async_copy_b128(dD + e * 16, delta + (size_t)(l0 + l) * DIN + c0 + q * 4);
        async_copy_b128(dX + e * 16, xsc   + (size_t)(l0 + l) * DIN + c0 + q * 4);
    }
    async_wait0();
    __syncthreads();

    const int cl = t >> 4, n = t & 15;
    float An = -__expf(Alog[(c0 + cl) * NST + n]);
    float a = 1.f, b = 0.f;
    for (int i = 0; i < SEGLEN; ++i) {
        float dlt = shD[i * 16 + cl];
        float dA  = __expf(dlt * An);
        float dBx = dlt * shX[i * 16 + cl] * shB[i * NST + n];
        a *= dA;
        b = b * dA + dBx;
    }
    int tid = blk * 256 + t;              // == seg*CN16 + c*16 + n
    aCar[tid] = a;
    bCar[tid] = b;
}

// ---------------- phase 2: exclusive scan of carries across segments ----------------
__global__ void scan_carry(const float* __restrict__ aCar, const float* __restrict__ bCar,
                           float* __restrict__ bIn) {
    int cn = blockIdx.x * blockDim.x + threadIdx.x;
    if (cn >= CN16) return;
    float cur = 0.f;
#pragma unroll
    for (int s = 0; s < SEGS; ++s) {
        bIn[(size_t)s * CN16 + cn] = cur;
        cur = cur * aCar[(size_t)s * CN16 + cn] + bCar[(size_t)s * CN16 + cn];
    }
}

// ---------------- phase 3: replay with carry-in, emit y ----------------
__global__ void __launch_bounds__(256)
scan_part2(const float* __restrict__ delta, const float* __restrict__ xsc,
           const float* __restrict__ Alog, const float* __restrict__ Bm,
           const float* __restrict__ Cm, const float* __restrict__ bIn,
           float* __restrict__ y) {
    __shared__ float shB[SEGLEN * NST];   // 16 KB
    __shared__ float shC[SEGLEN * NST];   // 16 KB
    __shared__ float shD[SEGLEN * 16];    // 16 KB
    __shared__ float shX[SEGLEN * 16];    // 16 KB
    const int t    = threadIdx.x;
    const int blk  = blockIdx.x;
    const int seg  = blk >> 7;
    const int cblk = blk & 127;
    const int c0   = cblk * 16;
    const int l0   = seg * SEGLEN;

    const unsigned dB = lds_off(shB), dC = lds_off(shC), dD = lds_off(shD), dX = lds_off(shX);
    const float* gB = Bm + (size_t)l0 * NST;
    const float* gC = Cm + (size_t)l0 * NST;
#pragma unroll
    for (int j = 0; j < 4; ++j) {
        int e = j * 256 + t;
        async_copy_b128(dB + e * 16, gB + e * 4);
        async_copy_b128(dC + e * 16, gC + e * 4);
        int l = e >> 2, q = e & 3;
        async_copy_b128(dD + e * 16, delta + (size_t)(l0 + l) * DIN + c0 + q * 4);
        async_copy_b128(dX + e * 16, xsc   + (size_t)(l0 + l) * DIN + c0 + q * 4);
    }
    async_wait0();
    __syncthreads();

    const int cl = t >> 4, n = t & 15;
    const int c = c0 + cl;
    float An = -__expf(Alog[c * NST + n]);
    int tid = blk * 256 + t;
    float st = bIn[tid];
    for (int i = 0; i < SEGLEN; ++i) {
        float dlt = shD[i * 16 + cl];
        float dA  = __expf(dlt * An);
        float dBx = dlt * shX[i * 16 + cl] * shB[i * NST + n];
        st = st * dA + dBx;
        float contrib = st * shC[i * NST + n];
#pragma unroll
        for (int m = 8; m > 0; m >>= 1) contrib += __shfl_xor(contrib, m, 32);
        if (n == 0) y[(size_t)(l0 + i) * DIN + c] = contrib;
    }
}

// ---------------- inner = (y + xs*D) * res  -> bf16 ----------------
__global__ void inner_kernel(const float* __restrict__ y, const float* __restrict__ xsc,
                             const float* __restrict__ Dv, const float* __restrict__ res,
                             __bf16* __restrict__ inner) {
    int idx = blockIdx.x * blockDim.x + threadIdx.x;
    if (idx >= LSEQ * DIN) return;
    int c = idx & (DIN - 1);
    inner[idx] = (__bf16)((y[idx] + xsc[idx] * Dv[c]) * res[idx]);
}

// ---------------- host orchestration ----------------
extern "C" void kernel_launch(void* const* d_in, const int* in_sizes, int n_in,
                              void* d_out, int out_size, void* d_ws, size_t ws_size,
                              hipStream_t stream) {
    const float* x       = (const float*)d_in[0];   // (2048,1024)
    const float* nscale  = (const float*)d_in[1];   // (1024,)
    const float* w_in    = (const float*)d_in[2];   // (1024,4096)
    const float* conv_w  = (const float*)d_in[3];   // (4,1,2048)
    const float* conv_b  = (const float*)d_in[4];   // (2048,)
    const float* A_log   = (const float*)d_in[5];   // (2048,16)
    const float* Dvec    = (const float*)d_in[6];   // (2048,)
    const float* w_xproj = (const float*)d_in[7];   // (2048,96)
    const float* w_dt    = (const float*)d_in[8];   // (64,2048)
    const float* b_dt    = (const float*)d_in[9];   // (2048,)
    const float* w_out   = (const float*)d_in[10];  // (2048,1024)
    float* out = (float*)d_out;                     // (2048,1024)

    char* ws = (char*)d_ws;
    size_t off = 0;
    auto alloc = [&](size_t bytes) -> void* {
        void* p = ws + off;
        off += (bytes + 255) & ~(size_t)255;
        return p;
    };

    __bf16* wInT    = (__bf16*)alloc((size_t)4096 * 1024 * 2);  // (N=4096, K=1024)
    __bf16* wOutT   = (__bf16*)alloc((size_t)1024 * 2048 * 2);  // (N=1024, K=2048)
    __bf16* wXprojT = (__bf16*)alloc((size_t)96   * 2048 * 2);  // (N=96,   K=2048)
    __bf16* wDtT    = (__bf16*)alloc((size_t)2048 * 64   * 2);  // (N=2048, K=64)
    __bf16* Hnorm   = (__bf16*)alloc((size_t)LSEQ * DMOD * 2);
    float*  xs_pre  = (float*) alloc((size_t)LSEQ * DIN * 4);
    float*  res     = (float*) alloc((size_t)LSEQ * DIN * 4);
    float*  xs_conv = (float*) alloc((size_t)LSEQ * DIN * 4);
    __bf16* xs_bf   = (__bf16*)alloc((size_t)LSEQ * DIN * 2);
    float*  proj    = (float*) alloc((size_t)LSEQ * 96 * 4);
    __bf16* dlow    = (__bf16*)alloc((size_t)LSEQ * DTRK * 2);
    float*  Bm      = (float*) alloc((size_t)LSEQ * NST * 4);
    float*  Cm      = (float*) alloc((size_t)LSEQ * NST * 4);
    float*  aCar    = (float*) alloc((size_t)SEGS * CN16 * 4);
    float*  bCar    = (float*) alloc((size_t)SEGS * CN16 * 4);
    float*  bIn     = (float*) alloc((size_t)SEGS * CN16 * 4);
    float*  ybuf    = (float*) alloc((size_t)LSEQ * DIN * 4);
    __bf16* inner   = (__bf16*)alloc((size_t)LSEQ * DIN * 2);
    float*  delta   = xs_pre;  // reuse: xs_pre dead after conv

    const int TB = 256;

    // 1) weight prepack (transpose + bf16)
    transpose_to_bf16<<<(1024 * 4096 + TB - 1) / TB, TB, 0, stream>>>(w_in, wInT, 1024, 4096);
    transpose_to_bf16<<<(2048 * 1024 + TB - 1) / TB, TB, 0, stream>>>(w_out, wOutT, 2048, 1024);
    transpose_to_bf16<<<(2048 * 96   + TB - 1) / TB, TB, 0, stream>>>(w_xproj, wXprojT, 2048, 96);
    transpose_to_bf16<<<(64 * 2048   + TB - 1) / TB, TB, 0, stream>>>(w_dt, wDtT, 64, 2048);

    // 2) RMSNorm
    rmsnorm_kernel<<<LSEQ, 256, 0, stream>>>(x, nscale, Hnorm);

    // 3) GEMM1: (2048x1024)x(1024x4096) -> xs_pre | silu(res)
    gemm_big_wmma<1><<<dim3(4096 / 128, LSEQ / 64), 128, 0, stream>>>(
        Hnorm, wInT, LSEQ, 4096, 1024, xs_pre, res, nullptr);

    // 4) causal depthwise conv + silu
    conv_silu_kernel<<<(LSEQ * DIN + TB - 1) / TB, TB, 0, stream>>>(
        xs_pre, conv_w, conv_b, xs_conv, xs_bf);

    // 5) proj = xs @ w_xproj : (2048x2048)x(2048x96)  (small-N kernel)
    gemm_bf16_wmma<0><<<dim3(96 / 32, LSEQ / 64), 128, 0, stream>>>(
        xs_bf, wXprojT, LSEQ, 96, 2048, proj, nullptr, nullptr);

    // 6) split proj
    split_proj_kernel<<<(LSEQ * 96 + TB - 1) / TB, TB, 0, stream>>>(proj, dlow, Bm, Cm);

    // 7) delta = softplus(dlow @ w_dt + b_dt) : (2048x64)x(64x2048)  [delta aliases xs_pre]
    gemm_big_wmma<2><<<dim3(2048 / 128, LSEQ / 64), 128, 0, stream>>>(
        dlow, wDtT, LSEQ, 2048, 64, delta, nullptr, b_dt);

    // 8) chunked selective scan (async-to-LDS staged)
    scan_part1<<<(SEGS * CN16) / TB, TB, 0, stream>>>(delta, xs_conv, A_log, Bm, aCar, bCar);
    scan_carry<<<CN16 / TB, TB, 0, stream>>>(aCar, bCar, bIn);
    scan_part2<<<(SEGS * CN16) / TB, TB, 0, stream>>>(delta, xs_conv, A_log, Bm, Cm, bIn, ybuf);

    // 9) inner = (y + xs*D) * res
    inner_kernel<<<(LSEQ * DIN + TB - 1) / TB, TB, 0, stream>>>(ybuf, xs_conv, Dvec, res, inner);

    // 10) out = x + inner @ w_out : (2048x2048)x(2048x1024)
    gemm_big_wmma<3><<<dim3(1024 / 128, LSEQ / 64), 128, 0, stream>>>(
        inner, wOutT, LSEQ, 1024, 2048, out, nullptr, x);

    (void)in_sizes; (void)n_in; (void)out_size; (void)ws_size;
}
